// DTransformer_6451040878724
// MI455X (gfx1250) — matrix operand order repaired and checked by reading, not compiled, not run on hardware
//
#include <hip/hip_runtime.h>

#define D_MODEL 256
#define N_HEADS 8
#define D_K 32
#define BSZ 2
#define SEQN 2048
#define MROWS (BSZ * SEQN)
#define SP (SEQN + 8)

typedef __attribute__((ext_vector_type(16))) _Float16 v16h;
typedef __attribute__((ext_vector_type(8)))  float    v8f;
typedef __attribute__((ext_vector_type(4)))  unsigned int v4u;
typedef __attribute__((ext_vector_type(8)))  int      v8i;
typedef __attribute__((ext_vector_type(4)))  int      v4i;

__device__ __forceinline__ v8f wmma16(v16h a, v16h b, v8f c) {
  // D = A(16x32 f16) * B(32x16 f16) + C(16x16 f32)
  return __builtin_amdgcn_wmma_f32_16x16x32_f16(false, a, false, b, (short)0, c,
                                                false, false);
}

// ---- fragment builders (CDNA5 wave32 WMMA VGPR layouts) ----
// A (16x32, MxK, 16-bit): lane m=lane&15, g=lane>>4;
//   elems 0..7 -> K = g*8 + i ; elems 8..15 -> K = 16 + g*8 + i
__device__ __forceinline__ v16h frag_a_f32(const float* base, int ld, int lane) {
  int m = lane & 15, g = lane >> 4;
  const float* p = base + (size_t)m * ld + g * 8;
  v16h r;
#pragma unroll
  for (int i = 0; i < 8; ++i) { r[i] = (_Float16)p[i]; r[8 + i] = (_Float16)p[16 + i]; }
  return r;
}
__device__ __forceinline__ v16h frag_a_f16(const _Float16* base, int ld, int lane) {
  int m = lane & 15, g = lane >> 4;
  const _Float16* p = base + (size_t)m * ld + g * 8;
  v16h r;
#pragma unroll
  for (int i = 0; i < 8; ++i) { r[i] = p[i]; r[8 + i] = p[16 + i]; }
  return r;
}
// B (32x16, KxN, 16-bit) built from K-major storage Bt[n][k]:
//   lane n=lane&15, g=lane>>4; elems 0..7 -> K = g*8 + i ; elems 8..15 -> K = 16 + g*8 + i
__device__ __forceinline__ v16h frag_b_f16(const _Float16* base, int ld, int lane) {
  int n = lane & 15, g = lane >> 4;
  const _Float16* p = base + (size_t)n * ld + g * 8;
  v16h r;
#pragma unroll
  for (int i = 0; i < 8; ++i) { r[i] = p[i]; r[8 + i] = p[16 + i]; }
  return r;
}

// ---- TDM: DMA a [rows x 256] fp32 tile into LDS with 8-dword row padding ----
// D# group0: count=1 | lds_addr | 57-bit global_addr | type=2
// D# group1: data_size=4B, pad_enable, pad_interval=256 dwords, pad_amount=8 dwords,
//            tensor_dim0=256, tensor_dim1=rows, tile_dim0=256, tile_dim1=rows,
//            tensor_dim0_stride=256
__device__ __forceinline__ void tdm_load_rows256(const float* gsrc, void* lds_dst,
                                                 int rows) {
  unsigned long long ga = (unsigned long long)(const void*)gsrc;
  unsigned int lds = (unsigned int)(size_t)lds_dst;
  v4u g0;
  g0[0] = 1u;                                        // count = 1 valid descriptor
  g0[1] = lds;                                       // lds_addr (bytes)
  g0[2] = (unsigned int)(ga & 0xffffffffu);          // global_addr[31:0]
  g0[3] = (unsigned int)((ga >> 32) & 0x1ffffffu) |  // global_addr[56:32]
          (2u << 30);                                // type = 2 ("image")
  v8i g1;
  g1[0] = (int)((2u << 16)      // data_size = 4 bytes
                | (1u << 20)    // pad_enable
                | (7u << 22)    // pad_interval: 256 dwords
                | (7u << 25));  // pad_amount: 8 dwords
  g1[1] = (int)(256u << 16);    // tensor_dim0 = 256 (bits 63:48)
  g1[2] = (int)((unsigned)rows << 16);  // tensor_dim1 (bits 111:80, low half)
  g1[3] = (int)(256u << 16);    // tile_dim0 = 256 (bits 127:112)
  g1[4] = rows;                 // tile_dim1 (bits 143:128)
  g1[5] = 256;                  // tensor_dim0_stride = 256 elements
  g1[6] = 0;
  g1[7] = 0;
  v4i gz4 = {0, 0, 0, 0};
  v8i gz8 = {0, 0, 0, 0, 0, 0, 0, 0};
  __builtin_amdgcn_tensor_load_to_lds(g0, g1, gz4, gz4, gz8, 0);
}

// ---------------- weight transpose + f16 convert: Wt[n][k] = W[k][n] ----------
__global__ void __launch_bounds__(256) k_wt(const float* __restrict__ W,
                                            _Float16* __restrict__ Wt) {
  int n = blockIdx.x, k = threadIdx.x;
  Wt[(size_t)n * D_MODEL + k] = (_Float16)W[(size_t)k * D_MODEL + n];
}

// ---------------- broadcast know_params row to [M,256] fp32 -------------------
__global__ void __launch_bounds__(256) k_know(const float* __restrict__ know,
                                              float* __restrict__ out) {
  for (size_t idx = (size_t)blockIdx.x * blockDim.x + threadIdx.x;
       idx < (size_t)MROWS * D_MODEL; idx += (size_t)gridDim.x * blockDim.x)
    out[idx] = know[idx & (D_MODEL - 1)];
}

// ---------------- GEMM over 16 rows x 256 cols per block ----------------------
// mode 0: out16 = (f16)(X*W + b)
// mode 1: out32 = layernorm(res + X*W + b) * g + bln
__global__ void __launch_bounds__(256) k_gemm_rows(
    const float* __restrict__ X, const _Float16* __restrict__ Wt,
    const float* __restrict__ bias, _Float16* __restrict__ out16,
    float* __restrict__ out32, const float* __restrict__ res,
    const float* __restrict__ lng, const float* __restrict__ lnb, int mode) {
  __shared__ float tile[16][D_MODEL + 8];
  int tid = threadIdx.x, lane = tid & 31, w = tid >> 5;
  int r0 = blockIdx.x * 16;
  const float* Xb = X + (size_t)r0 * D_MODEL;

  // stage the shared A tile once per workgroup via the Tensor Data Mover
  if (w == 0) {
    tdm_load_rows256(Xb, &tile[0][0], 16);
    __builtin_amdgcn_s_wait_tensorcnt(0);
  }
  __syncthreads();

  int n0 = w * 32;
  int mA = lane & 15, gA = lane >> 4;
  v8f acc0 = {}, acc1 = {};
#pragma unroll
  for (int k0 = 0; k0 < D_MODEL; k0 += 32) {
    v16h a;
    {
      const float* pa = &tile[mA][k0 + gA * 8];
#pragma unroll
      for (int i = 0; i < 8; ++i) {
        a[i] = (_Float16)pa[i];
        a[8 + i] = (_Float16)pa[16 + i];
      }
    }
    v16h b0 = frag_b_f16(Wt + (size_t)n0 * D_MODEL + k0, D_MODEL, lane);
    v16h b1 = frag_b_f16(Wt + (size_t)(n0 + 16) * D_MODEL + k0, D_MODEL, lane);
    acc0 = wmma16(a, b0, acc0);
    acc1 = wmma16(a, b1, acc1);
  }
  int nn = lane & 15, g = lane >> 4;
  if (mode == 0) {
#pragma unroll
    for (int j = 0; j < 8; ++j) {
      int m = j + 8 * g;
      int c0 = n0 + nn, c1 = n0 + 16 + nn;
      out16[(size_t)(r0 + m) * D_MODEL + c0] = (_Float16)(acc0[j] + bias[c0]);
      out16[(size_t)(r0 + m) * D_MODEL + c1] = (_Float16)(acc1[j] + bias[c1]);
    }
  } else {
    __syncthreads();  // everyone finished reading A tile before we overwrite it
#pragma unroll
    for (int j = 0; j < 8; ++j) {
      int m = j + 8 * g;
      int c0 = n0 + nn, c1 = n0 + 16 + nn;
      tile[m][c0] = acc0[j] + bias[c0] + res[(size_t)(r0 + m) * D_MODEL + c0];
      tile[m][c1] = acc1[j] + bias[c1] + res[(size_t)(r0 + m) * D_MODEL + c1];
    }
    __syncthreads();
    for (int rr = 0; rr < 2; ++rr) {
      int m = w * 2 + rr;
      float s = 0.f, s2 = 0.f;
      for (int c = lane; c < D_MODEL; c += 32) {
        float x = tile[m][c];
        s += x; s2 += x * x;
      }
      for (int off = 16; off; off >>= 1) {
        s += __shfl_xor(s, off);
        s2 += __shfl_xor(s2, off);
      }
      float mean = s * (1.f / D_MODEL);
      float var = s2 * (1.f / D_MODEL) - mean * mean;
      float rstd = rsqrtf(var + 1e-5f);
      for (int c = lane; c < D_MODEL; c += 32) {
        float x = (tile[m][c] - mean) * rstd * lng[c] + lnb[c];
        out32[(size_t)(r0 + m) * D_MODEL + c] = x;
      }
    }
  }
}

// ---------------- attention with distance-decay, one block per (b,h,qtile) ----
__global__ void __launch_bounds__(256) k_attn(
    const _Float16* __restrict__ Q, const _Float16* __restrict__ K,
    const _Float16* __restrict__ V, const float* __restrict__ gam,
    float* __restrict__ out, int peek) {
  extern __shared__ char smem[];
  float* sc = (float*)smem;                                   // [16][SP]
  _Float16* vt = (_Float16*)(smem + (size_t)16 * SP * 4);     // [8][16][32]
  float* part = (float*)(smem + (size_t)16 * SP * 4 + 8 * 16 * 32 * 2); // [8][16][16]

  int tid = threadIdx.x, lane = tid & 31, w = tid >> 5;
  int qt = blockIdx.x & 127;
  int h = (blockIdx.x >> 7) & 7;
  int b = blockIdx.x >> 10;
  size_t rowbase = (size_t)b * SEQN;
  int col = h * D_K;
  int nn = lane & 15, g = lane >> 4;
  const float scale = 0.17677669529663687f;  // 1/sqrt(32)

  // ---- phase 1: scores = Q * K^T / sqrt(dk), strip resident in LDS ----
  v16h aq = frag_a_f16(Q + (rowbase + qt * 16) * D_MODEL + col, D_MODEL, lane);
  for (int kt = w; kt < SEQN / 16; kt += 8) {
    if (kt + 8 < SEQN / 16)
      __builtin_prefetch(K + (rowbase + (kt + 8) * 16) * D_MODEL + col, 0, 0);
    v16h bk = frag_b_f16(K + (rowbase + kt * 16) * D_MODEL + col, D_MODEL, lane);
    v8f s = {};
    s = wmma16(aq, bk, s);
#pragma unroll
    for (int j = 0; j < 8; ++j)
      sc[(size_t)(j + 8 * g) * SP + kt * 16 + nn] = s[j] * scale;
  }
  __syncthreads();

  // ---- phase 2: masked softmax -> cumsum decay -> second softmax (per row) ----
  float gamma;
  {
    float x = gam[h];
    float sp = (x > 20.f) ? x : log1pf(__expf(x));
    gamma = -sp;
  }
  for (int rr = 0; rr < 2; ++rr) {
    int m = w * 2 + rr;
    float* row = sc + (size_t)m * SP;
    int qi = qt * 16 + m;
    int L = peek ? (qi + 1) : qi;  // number of valid keys
    if (L == 0) {
      for (int j = lane; j < SEQN; j += 32) row[j] = 0.f;
      continue;
    }
    float mx = -3.4e38f;
    for (int j = lane; j < L; j += 32) mx = fmaxf(mx, row[j]);
    for (int off = 16; off; off >>= 1) mx = fmaxf(mx, __shfl_xor(mx, off));
    float sum1 = 0.f;
    for (int j = lane; j < L; j += 32) sum1 += __expf(row[j] - mx);
    for (int off = 16; off; off >>= 1) sum1 += __shfl_xor(sum1, off);
    float inv1 = 1.f / sum1;

    float carry = 0.f, mx2 = -3.4e38f;
    for (int c = 0; c < SEQN; c += 32) {
      int j = c + lane;
      float s = row[j];
      float v = (j < L) ? __expf(s - mx) * inv1 : 0.f;
      float x = v;
#pragma unroll
      for (int off = 1; off < 32; off <<= 1) {
        float y = __shfl_up(x, off);
        if (lane >= off) x += y;
      }
      float incl = carry + x;          // inclusive cumsum of softmax
      carry += __shfl(x, 31);
      if (j < L) {
        float R = 1.f - incl;          // disttot - distcum
        float pd = fabsf((float)(qi - j));
        float dist = sqrtf(fmaxf(R * pd, 0.f));
        float eff = __expf(gamma * dist);
        eff = fminf(fmaxf(eff, 1e-5f), 1e5f);
        float s2 = s * eff;
        row[j] = s2;
        mx2 = fmaxf(mx2, s2);
      } else {
        row[j] = 0.f;
      }
    }
    for (int off = 16; off; off >>= 1) mx2 = fmaxf(mx2, __shfl_xor(mx2, off));
    float sum2 = 0.f;
    for (int j = lane; j < L; j += 32) sum2 += __expf(row[j] - mx2);
    for (int off = 16; off; off >>= 1) sum2 += __shfl_xor(sum2, off);
    float inv2 = 1.f / sum2;
    for (int j = lane; j < SEQN; j += 32)
      row[j] = (j < L) ? __expf(row[j] - mx2) * inv2 : 0.f;
  }
  __syncthreads();

  // ---- phase 3: O = P * V ; waves split (ntile in dk) x (4 K-slices) ----
  int ntile = w & 1, ks = w >> 1;
  _Float16* vtw = vt + (size_t)w * (16 * 32);
  v8f acc = {};
  for (int t = 0; t < 16; ++t) {
    int k0 = (ks * 16 + t) * 32;  // key base
    {  // stage V^T slice: vtw[n][k] = V[k0+k][col + ntile*16 + n]
      const _Float16* vp = V + (rowbase + k0 + lane) * D_MODEL + col + ntile * 16;
#pragma unroll
      for (int i = 0; i < 16; ++i) vtw[i * 32 + lane] = vp[i];
    }
    __syncthreads();  // uniform across all waves (same trip count)
    v16h a;
    {  // A from probs (fp32 in LDS)
      const float* p = sc + (size_t)nn * SP + k0 + g * 8;
#pragma unroll
      for (int i = 0; i < 8; ++i) { a[i] = (_Float16)p[i]; a[8 + i] = (_Float16)p[16 + i]; }
    }
    v16h bv = frag_b_f16(vtw, 32, lane);
    acc = wmma16(a, bv, acc);
    __syncthreads();
  }
#pragma unroll
  for (int j = 0; j < 8; ++j) part[w * 256 + (j + 8 * g) * 16 + nn] = acc[j];
  __syncthreads();
  for (int e = tid; e < 512; e += 256) {
    int nt = e >> 8;
    int pos = e & 255;
    float sum = 0.f;
#pragma unroll
    for (int s4 = 0; s4 < 4; ++s4) sum += part[(s4 * 2 + nt) * 256 + pos];
    int m = pos >> 4, n = pos & 15;
    out[(rowbase + qt * 16 + m) * D_MODEL + col + nt * 16 + n] = sum;
  }
}

// ---------------- key = sigmoid(know_chunk_h . lk_W + lk_b), shared by all n --
__global__ void __launch_bounds__(256) k_key(const float* __restrict__ know,
                                             const float* __restrict__ lkW,
                                             const float* __restrict__ lkb,
                                             float* __restrict__ keym) {
  int c = threadIdx.x;
  for (int h = 0; h < N_HEADS; ++h) {
    float acc = lkb[c];
#pragma unroll
    for (int k = 0; k < D_K; ++k) acc += know[h * D_K + k] * lkW[(size_t)k * D_MODEL + c];
    keym[(size_t)h * D_MODEL + c] = 1.f / (1.f + __expf(-acc));
  }
}

// ---------------- final: beta/alpha head mixing with sigmoid values ----------
__global__ void __launch_bounds__(256) k_final(
    const float* __restrict__ keym, const float* __restrict__ h3,
    const float* __restrict__ qe, const float* __restrict__ lvW,
    const float* __restrict__ lvb, float* __restrict__ out) {
  __shared__ float beta[8];
  int n = blockIdx.x, tid = threadIdx.x, lane = tid & 31, w = tid >> 5;
  const float* q = qe + (size_t)n * D_MODEL;
  float p = 0.f;
  for (int c = lane; c < D_MODEL; c += 32) p += keym[(size_t)w * D_MODEL + c] * q[c];
  for (int off = 16; off; off >>= 1) p += __shfl_xor(p, off);
  if (lane == 0) beta[w] = p;
  __syncthreads();
  float bm = beta[0];
#pragma unroll
  for (int i = 1; i < 8; ++i) bm = fmaxf(bm, beta[i]);
  float al[8], den = 0.f;
#pragma unroll
  for (int i = 0; i < 8; ++i) { al[i] = __expf(beta[i] - bm); den += al[i]; }
  float invd = 1.f / den;
  int c = tid;
  const float* hr = h3 + (size_t)n * D_MODEL;
  float o = 0.f;
#pragma unroll
  for (int hh = 0; hh < N_HEADS; ++hh) {
    float acc = lvb[c];
#pragma unroll
    for (int k = 0; k < D_K; ++k) acc += hr[hh * D_K + k] * lvW[(size_t)k * D_MODEL + c];
    float v = 1.f / (1.f + __expf(-acc));
    o += (al[hh] * invd) * v;
  }
  out[(size_t)n * D_MODEL + c] = o;
}

// =============================================================================
extern "C" void kernel_launch(void* const* d_in, const int* in_sizes, int n_in,
                              void* d_out, int out_size, void* d_ws, size_t ws_size,
                              hipStream_t stream) {
  const float* q_emb = (const float*)d_in[0];
  const float* qa_emb = (const float*)d_in[1];
  const float* Wsrc[10] = {(const float*)d_in[2],  (const float*)d_in[4],
                           (const float*)d_in[6],  (const float*)d_in[11],
                           (const float*)d_in[13], (const float*)d_in[15],
                           (const float*)d_in[20], (const float*)d_in[22],
                           (const float*)d_in[24], (const float*)d_in[26]};
  const float* b1_bq = (const float*)d_in[3];
  const float* b1_bv = (const float*)d_in[5];
  const float* b1_bo = (const float*)d_in[7];
  const float* b1_gam = (const float*)d_in[8];
  const float* b1_lng = (const float*)d_in[9];
  const float* b1_lnb = (const float*)d_in[10];
  const float* b2_bq = (const float*)d_in[12];
  const float* b2_bv = (const float*)d_in[14];
  const float* b2_bo = (const float*)d_in[16];
  const float* b2_gam = (const float*)d_in[17];
  const float* b2_lng = (const float*)d_in[18];
  const float* b2_lnb = (const float*)d_in[19];
  const float* b3_bq = (const float*)d_in[21];
  const float* b3_bk = (const float*)d_in[23];
  const float* b3_bv = (const float*)d_in[25];
  const float* b3_bo = (const float*)d_in[27];
  const float* b3_gam = (const float*)d_in[28];
  const float* b3_lng = (const float*)d_in[29];
  const float* b3_lnb = (const float*)d_in[30];
  const float* know = (const float*)d_in[31];
  const float* lk_W = (const float*)d_in[32];
  const float* lk_b = (const float*)d_in[33];
  const float* lv_W = (const float*)d_in[34];
  const float* lv_b = (const float*)d_in[35];

  // ---- workspace carve ----
  char* p = (char*)d_ws;
  auto alloc = [&](size_t bytes) {
    void* r = (void*)p;
    p += (bytes + 255) & ~(size_t)255;
    return r;
  };
  _Float16* WT[10];
  for (int i = 0; i < 10; ++i) WT[i] = (_Float16*)alloc((size_t)D_MODEL * D_MODEL * 2);
  _Float16* F16Q = (_Float16*)alloc((size_t)MROWS * D_MODEL * 2);
  _Float16* F16K = (_Float16*)alloc((size_t)MROWS * D_MODEL * 2);
  _Float16* F16V = (_Float16*)alloc((size_t)MROWS * D_MODEL * 2);
  float* ATT = (float*)alloc((size_t)MROWS * D_MODEL * 4);
  float* HQ = (float*)alloc((size_t)MROWS * D_MODEL * 4);
  float* HA = (float*)alloc((size_t)MROWS * D_MODEL * 4);
  float* KNOWB = (float*)alloc((size_t)MROWS * D_MODEL * 4);
  float* H3 = (float*)alloc((size_t)MROWS * D_MODEL * 4);
  float* KEYM = (float*)alloc((size_t)N_HEADS * D_MODEL * 4);

  const int gemm_grid = MROWS / 16;
  const size_t attn_lds = (size_t)16 * SP * 4 + 8 * 16 * 32 * 2 + 8 * 256 * 4;

  for (int i = 0; i < 10; ++i)
    k_wt<<<D_MODEL, D_MODEL, 0, stream>>>(Wsrc[i], WT[i]);
  k_know<<<1024, 256, 0, stream>>>(know, KNOWB);

  // ---- block 1 (kq_same: K projection == Q projection) ----
  k_gemm_rows<<<gemm_grid, 256, 0, stream>>>(q_emb, WT[0], b1_bq, F16Q, nullptr,
                                             nullptr, nullptr, nullptr, 0);
  k_gemm_rows<<<gemm_grid, 256, 0, stream>>>(q_emb, WT[1], b1_bv, F16V, nullptr,
                                             nullptr, nullptr, nullptr, 0);
  k_attn<<<BSZ * N_HEADS * (SEQN / 16), 256, attn_lds, stream>>>(F16Q, F16Q, F16V,
                                                                 b1_gam, ATT, 1);
  k_gemm_rows<<<gemm_grid, 256, 0, stream>>>(ATT, WT[2], b1_bo, nullptr, HQ,
                                             q_emb, b1_lng, b1_lnb, 1);
  // ---- block 2 ----
  k_gemm_rows<<<gemm_grid, 256, 0, stream>>>(qa_emb, WT[3], b2_bq, F16Q, nullptr,
                                             nullptr, nullptr, nullptr, 0);
  k_gemm_rows<<<gemm_grid, 256, 0, stream>>>(qa_emb, WT[4], b2_bv, F16V, nullptr,
                                             nullptr, nullptr, nullptr, 0);
  k_attn<<<BSZ * N_HEADS * (SEQN / 16), 256, attn_lds, stream>>>(F16Q, F16Q, F16V,
                                                                 b2_gam, ATT, 1);
  k_gemm_rows<<<gemm_grid, 256, 0, stream>>>(ATT, WT[5], b2_bo, nullptr, HA,
                                             qa_emb, b2_lng, b2_lnb, 1);
  // ---- block 3 (query = broadcast(know), strict causal mask) ----
  k_gemm_rows<<<gemm_grid, 256, 0, stream>>>(KNOWB, WT[6], b3_bq, F16Q, nullptr,
                                             nullptr, nullptr, nullptr, 0);
  k_gemm_rows<<<gemm_grid, 256, 0, stream>>>(HQ, WT[7], b3_bk, F16K, nullptr,
                                             nullptr, nullptr, nullptr, 0);
  k_gemm_rows<<<gemm_grid, 256, 0, stream>>>(HA, WT[8], b3_bv, F16V, nullptr,
                                             nullptr, nullptr, nullptr, 0);
  k_attn<<<BSZ * N_HEADS * (SEQN / 16), 256, attn_lds, stream>>>(F16Q, F16K, F16V,
                                                                 b3_gam, ATT, 0);
  k_gemm_rows<<<gemm_grid, 256, 0, stream>>>(ATT, WT[9], b3_bo, nullptr, H3,
                                             KNOWB, b3_lng, b3_lnb, 1);
  // ---- final head-mixing ----
  k_key<<<1, 256, 0, stream>>>(know, lk_W, lk_b, KEYM);
  k_final<<<MROWS, 256, 0, stream>>>(KEYM, H3, q_emb, lv_W, lv_b, (float*)d_out);
}
